// TransformerLayer_37434934952831
// MI455X (gfx1250) — compile-verified, matrix operand order checked by reading, not run on hardware
//
#include <hip/hip_runtime.h>

// ---------------------------------------------------------------------------
// TransformerLayer forward for MI455X (gfx1250, wave32, WMMA 16x16x32 f16)
// B=8, S=1024, D=768, H=12, DH=64
// ---------------------------------------------------------------------------

typedef _Float16 half_t;
typedef __attribute__((ext_vector_type(16))) _Float16 v16h;
typedef __attribute__((ext_vector_type(8)))  float    v8f;
typedef int v4i __attribute__((vector_size(16)));

#define BB   8
#define SS   1024
#define DD   768
#define HH   12
#define DHH  64
#define MTOT (BB * SS)      // 8192
#define DFF  (2 * DD)       // 1536

// ---- gfx1250 async global->LDS path (guarded; falls back cleanly) --------
#if defined(__HIP_DEVICE_COMPILE__) && \
    __has_builtin(__builtin_amdgcn_global_load_async_to_lds_b128) && \
    __has_builtin(__builtin_amdgcn_s_wait_asynccnt)
#define HAVE_ASYNC_LDS 1
#endif

typedef __attribute__((address_space(1))) v4i* gvec_p;   // global v4i*
typedef __attribute__((address_space(3))) v4i* lvec_p;   // LDS v4i*

__device__ __forceinline__ void stage16(const half_t* gsrc, half_t* ldst) {
#ifdef HAVE_ASYNC_LDS
  __builtin_amdgcn_global_load_async_to_lds_b128(
      (gvec_p)(unsigned long long)(gsrc),
      (lvec_p)(unsigned long long)(ldst), 0, 0);
#else
  *(uint4*)ldst = *(const uint4*)gsrc;
#endif
}

__device__ __forceinline__ void stage_wait() {
#ifdef HAVE_ASYNC_LDS
  __builtin_amdgcn_s_wait_asynccnt(0);
#endif
}

// ---------------------------------------------------------------------------
// Fragment loader: 16x32 tile, element (m, kk) at base[(row0+m)*stride + k0+kk]
// ISA 16-bit A layout: lane l holds row m=l&15; lanes 0-15 K=0..7,16..23;
// lanes 16-31 K=8..15,24..31.  B-fragment uses the same pattern with the
// "row" being the output column (weights pre-transposed to N x K).
// ---------------------------------------------------------------------------
__device__ __forceinline__ v16h load_frag(const half_t* __restrict__ base,
                                          int stride, int row0, int k0, int lane) {
  const half_t* p = base + (size_t)(row0 + (lane & 15)) * stride
                         + k0 + ((lane >> 4) * 8);
  v16h f;
#pragma unroll
  for (int i = 0; i < 8; ++i) f[i]     = p[i];
#pragma unroll
  for (int i = 0; i < 8; ++i) f[8 + i] = p[16 + i];
  return f;
}

// ---------------------------------------------------------------------------
// Elementwise conversions
// ---------------------------------------------------------------------------
__global__ void cvt_f32_to_f16(const float* __restrict__ in,
                               half_t* __restrict__ out, int n) {
  int i = blockIdx.x * blockDim.x + threadIdx.x;
  if (i < n) out[i] = (half_t)in[i];
}

// W is [K x N] row-major fp32 (y = x @ W).  Produce Wt [N x K] f16 so the
// WMMA B-fragment (column of W) is a contiguous K-run.
__global__ void transpose_cvt(const float* __restrict__ W,
                              half_t* __restrict__ Wt, int K, int N) {
  int i = blockIdx.x * blockDim.x + threadIdx.x;
  if (i < K * N) {
    int k = i / N, n = i - k * N;
    Wt[(size_t)n * K + k] = (half_t)W[i];
  }
}

// ---------------------------------------------------------------------------
// WMMA GEMM:  C[M x N] = A[M x K] * Bt[N x K]^T  (+ epilogue)
// Block = 128 threads (4 waves); block tile = 16(M) x 256(N).
// The 16x32 A tile is staged into LDS once per K-step (async-to-LDS,
// double-buffered) and reused by all 4 waves; each wave computes a 16x64
// strip (4 accumulators) so each A fragment feeds 4 WMMAs.
//   mode 0: outH = f16(acc + bias)                      (Q, K projections)
//   mode 1: outH = f16(acc + bias), stored transposed   (V -> [B,H,DH,S])
//   mode 2: outF = acc + bias + res                     (Wo+residual, FFN2+h)
//   mode 3: outH = f16(relu(acc + bias))                (FFN1)
// ---------------------------------------------------------------------------
__global__ void __launch_bounds__(128)
gemm_wmma(const half_t* __restrict__ A, const half_t* __restrict__ Bt,
          const float* __restrict__ bias, const float* __restrict__ res,
          float* __restrict__ outF, half_t* __restrict__ outH,
          int M, int N, int K, int mode) {
  __shared__ half_t Asm[2][16 * 32];

  const int tid  = threadIdx.x;
  const int lane = tid & 31;
  const int wave = tid >> 5;
  const int tm   = blockIdx.y * 16;               // output row tile
  const int tn0  = blockIdx.x * 256 + wave * 64;  // this wave's first column

  // Stage one 16x32 A tile (1 KB): 64 x b128 transfers by the first 2 waves.
  auto stage_a = [&](int buf, int k) {
    if (tid < 64) {
      const int r = tid >> 2;            // row 0..15
      const int c = (tid & 3) * 8;       // 8-half chunk
      stage16(&A[(size_t)(tm + r) * K + k + c], &Asm[buf][r * 32 + c]);
    }
  };

  stage_a(0, 0);
  stage_wait();
  __syncthreads();

  v8f acc[4] = {v8f{}, v8f{}, v8f{}, v8f{}};
  int buf = 0;
  for (int k = 0; k < K; k += 32) {
    const bool more = (k + 32) < K;
    if (more) {
      stage_a(buf ^ 1, k + 32);          // async: overlaps with compute below
      __builtin_prefetch(&Bt[(size_t)(tn0 + (lane & 15)) * K + k + 32], 0, 1);
    }

    // B fragments for 4 column tiles (global, K-contiguous)
    v16h b0 = load_frag(Bt, K, tn0 +  0, k, lane);
    v16h b1 = load_frag(Bt, K, tn0 + 16, k, lane);
    v16h b2 = load_frag(Bt, K, tn0 + 32, k, lane);
    v16h b3 = load_frag(Bt, K, tn0 + 48, k, lane);
    // Shared A fragment from LDS
    v16h a  = load_frag(Asm[buf], 32, 0, 0, lane);

    acc[0] = __builtin_amdgcn_wmma_f32_16x16x32_f16(false, a, false, b0, (short)0, acc[0], false, false);
    acc[1] = __builtin_amdgcn_wmma_f32_16x16x32_f16(false, a, false, b1, (short)0, acc[1], false, false);
    acc[2] = __builtin_amdgcn_wmma_f32_16x16x32_f16(false, a, false, b2, (short)0, acc[2], false, false);
    acc[3] = __builtin_amdgcn_wmma_f32_16x16x32_f16(false, a, false, b3, (short)0, acc[3], false, false);

    if (more) {
      stage_wait();      // per-wave async complete
      __syncthreads();   // all waves' transfers visible
    }
    buf ^= 1;
  }

  const int col  = lane & 15;
  const int half = lane >> 4;

#pragma unroll
  for (int j = 0; j < 4; ++j) {
    const int gn = tn0 + j * 16 + col;
    const float bv = bias[gn];
#pragma unroll
    for (int r = 0; r < 8; ++r) {
      const int gm = tm + r + half * 8;        // C layout: VGPR r -> M = r + 8*half
      float v = acc[j][r] + bv;
      if (mode == 0) {
        outH[(size_t)gm * N + gn] = (half_t)v;
      } else if (mode == 1) {                  // V transposed: [B][H][DH][S]
        int b_ = gm >> 10, s_ = gm & 1023;
        int h_ = gn >> 6,  d_ = gn & 63;
        outH[((size_t)((b_ * HH + h_) * DHH + d_)) * SS + s_] = (half_t)v;
      } else if (mode == 2) {
        outF[(size_t)gm * N + gn] = v + res[(size_t)gm * N + gn];
      } else {                                 // mode 3: ReLU
        outH[(size_t)gm * N + gn] = (half_t)fmaxf(v, 0.0f);
      }
    }
  }
}

// ---------------------------------------------------------------------------
// Attention: one wave per 16-query tile of one (b, h).
//   scores(16xS) -> LDS fp32 (scaled + masked) -> softmax -> probs f16 in LDS
//   -> ctx(16x64) via WMMA over S -> normalize -> ctx buffer [M x D] f16
// Dynamic LDS: 16*1024*4 (scores) + 16*1024*2 (probs) + 64 (1/rowsum)
// ---------------------------------------------------------------------------
__global__ void __launch_bounds__(32)
attn_wmma(const half_t* __restrict__ Qh, const half_t* __restrict__ Kh,
          const half_t* __restrict__ Vt, const int* __restrict__ mask,
          half_t* __restrict__ ctx) {
  extern __shared__ char smem_raw[];
  float*  sc   = (float*)smem_raw;                                  // [16][1024]
  half_t* pr   = (half_t*)(smem_raw + 16 * SS * 4);                 // [16][1024]
  float*  rrow = (float*)(smem_raw + 16 * SS * 4 + 16 * SS * 2);    // [16]

  const int lane = threadIdx.x & 31;
  const int flat = blockIdx.x;            // B * H * (S/16) = 6144
  const int qt   = flat & 63;
  const int rest = flat >> 6;
  const int h    = rest % HH;
  const int b    = rest / HH;
  const int qrow0 = b * SS + qt * 16;

  // Q A-fragments (head slice, DH=64 -> two 32-wide K steps)
  const v16h qa0 = load_frag(Qh, DD, qrow0, h * DHH,      lane);
  const v16h qa1 = load_frag(Qh, DD, qrow0, h * DHH + 32, lane);

  const int col  = lane & 15;
  const int half = lane >> 4;

  // ---- Pass 1: scores = (Q K^T) * 1/8 + mask, into LDS --------------------
  for (int jt = 0; jt < SS / 16; ++jt) {
    const int krow0 = b * SS + jt * 16;       // B-fragment "rows" are keys
    v16h kb0 = load_frag(Kh, DD, krow0, h * DHH,      lane);
    v16h kb1 = load_frag(Kh, DD, krow0, h * DHH + 32, lane);
    v8f c = {};
    c = __builtin_amdgcn_wmma_f32_16x16x32_f16(false, qa0, false, kb0,
                                               (short)0, c, false, false);
    c = __builtin_amdgcn_wmma_f32_16x16x32_f16(false, qa1, false, kb1,
                                               (short)0, c, false, false);
    const int key  = jt * 16 + col;
    const float ma = mask[b * SS + key] ? 0.0f : -1e9f;
#pragma unroll
    for (int r = 0; r < 8; ++r)
      sc[(r + half * 8) * SS + key] = c[r] * 0.125f + ma;
  }
  __syncthreads();

  // ---- Pass 2: softmax (2 lanes per row, 512 cols each) -------------------
  {
    const int row  = lane & 15;
    const int base = (lane >> 4) * 512;
    float m = -1e30f;
    for (int j = 0; j < 512; ++j) m = fmaxf(m, sc[row * SS + base + j]);
    m = fmaxf(m, __shfl_xor(m, 16, 32));
    float s = 0.0f;
    for (int j = 0; j < 512; ++j) {
      float e = __expf(sc[row * SS + base + j] - m);
      s += e;
      pr[row * SS + base + j] = (half_t)e;   // unnormalized probs, f16
    }
    s += __shfl_xor(s, 16, 32);
    if (lane < 16) rrow[lane] = 1.0f / s;
  }
  __syncthreads();

  // ---- Pass 3: ctx = P @ V  (16 x 64), WMMA over S ------------------------
  v8f c0 = {}, c1 = {}, c2 = {}, c3 = {};
  const int vrow0 = (b * HH + h) * DHH;      // Vt rows = head features
  for (int kt = 0; kt < SS; kt += 32) {
    v16h pa = load_frag(pr, SS, 0, kt, lane);             // probs from LDS
    v16h v0 = load_frag(Vt, SS, vrow0 +  0, kt, lane);
    v16h v1 = load_frag(Vt, SS, vrow0 + 16, kt, lane);
    v16h v2 = load_frag(Vt, SS, vrow0 + 32, kt, lane);
    v16h v3 = load_frag(Vt, SS, vrow0 + 48, kt, lane);
    c0 = __builtin_amdgcn_wmma_f32_16x16x32_f16(false, pa, false, v0, (short)0, c0, false, false);
    c1 = __builtin_amdgcn_wmma_f32_16x16x32_f16(false, pa, false, v1, (short)0, c1, false, false);
    c2 = __builtin_amdgcn_wmma_f32_16x16x32_f16(false, pa, false, v2, (short)0, c2, false, false);
    c3 = __builtin_amdgcn_wmma_f32_16x16x32_f16(false, pa, false, v3, (short)0, c3, false, false);
  }

#pragma unroll
  for (int r = 0; r < 8; ++r) {
    const int row = r + half * 8;
    const float rs = rrow[row];
    half_t* o = ctx + (size_t)(qrow0 + row) * DD + h * DHH + col;
    o[0]  = (half_t)(c0[r] * rs);
    o[16] = (half_t)(c1[r] * rs);
    o[32] = (half_t)(c2[r] * rs);
    o[48] = (half_t)(c3[r] * rs);
  }
}

// ---------------------------------------------------------------------------
// LayerNorm over D=768; one block (256 thr) per row.  Writes fp32 (+opt f16).
// ---------------------------------------------------------------------------
__global__ void __launch_bounds__(256)
layernorm(const float* __restrict__ in, const float* __restrict__ g,
          const float* __restrict__ be, float* __restrict__ outF,
          half_t* __restrict__ outH) {
  __shared__ float red[256];
  const int row = blockIdx.x;
  const int t   = threadIdx.x;
  const float* x = in + (size_t)row * DD;

  float s = 0.0f;
  for (int i = t; i < DD; i += 256) s += x[i];
  red[t] = s; __syncthreads();
  for (int off = 128; off > 0; off >>= 1) {
    if (t < off) red[t] += red[t + off];
    __syncthreads();
  }
  const float mu = red[0] * (1.0f / DD);
  __syncthreads();

  float v = 0.0f;
  for (int i = t; i < DD; i += 256) { float d = x[i] - mu; v += d * d; }
  red[t] = v; __syncthreads();
  for (int off = 128; off > 0; off >>= 1) {
    if (t < off) red[t] += red[t + off];
    __syncthreads();
  }
  const float rstd = rsqrtf(red[0] * (1.0f / DD) + 1e-5f);

  for (int i = t; i < DD; i += 256) {
    float y = (x[i] - mu) * rstd * g[i] + be[i];
    outF[(size_t)row * DD + i] = y;
    if (outH) outH[(size_t)row * DD + i] = (half_t)y;
  }
}

// ---------------------------------------------------------------------------
// Host-side orchestration
// ---------------------------------------------------------------------------
extern "C" void kernel_launch(void* const* d_in, const int* in_sizes, int n_in,
                              void* d_out, int out_size, void* d_ws, size_t ws_size,
                              hipStream_t stream) {
  const float* x   = (const float*)d_in[0];
  const int*   msk = (const int*)  d_in[1];
  const float* Wq  = (const float*)d_in[2];
  const float* bq  = (const float*)d_in[3];
  const float* Wk  = (const float*)d_in[4];
  const float* bk  = (const float*)d_in[5];
  const float* Wv  = (const float*)d_in[6];
  const float* bv  = (const float*)d_in[7];
  const float* Wo  = (const float*)d_in[8];
  const float* bo  = (const float*)d_in[9];
  const float* g1  = (const float*)d_in[10];
  const float* be1 = (const float*)d_in[11];
  const float* W1  = (const float*)d_in[12];
  const float* b1  = (const float*)d_in[13];
  const float* W2  = (const float*)d_in[14];
  const float* b2  = (const float*)d_in[15];
  const float* g2  = (const float*)d_in[16];
  const float* be2 = (const float*)d_in[17];
  float* out = (float*)d_out;

  char* ws = (char*)d_ws;
  size_t off = 0;
  auto alloc = [&](size_t bytes) -> char* {
    char* p = ws + off;
    off = (off + bytes + 255) & ~(size_t)255;
    return p;
  };

  const size_t MD  = (size_t)MTOT * DD;   // 6,291,456
  const size_t MF  = (size_t)MTOT * DFF;  // 12,582,912

  half_t* xh   = (half_t*)alloc(MD * 2);
  half_t* Wqt  = (half_t*)alloc((size_t)DD * DD * 2);
  half_t* Wkt  = (half_t*)alloc((size_t)DD * DD * 2);
  half_t* Wvt  = (half_t*)alloc((size_t)DD * DD * 2);
  half_t* Wot  = (half_t*)alloc((size_t)DD * DD * 2);
  half_t* W1t  = (half_t*)alloc((size_t)DD * DFF * 2);
  half_t* W2t  = (half_t*)alloc((size_t)DFF * DD * 2);
  half_t* Qh   = (half_t*)alloc(MD * 2);
  half_t* Kh   = (half_t*)alloc(MD * 2);
  half_t* Vt   = (half_t*)alloc(MD * 2);   // [B][H][DH][S]
  float*  attnF= (float*) alloc(MD * 4);
  float*  hF   = (float*) alloc(MD * 4);
  half_t* ff1h = (half_t*)alloc(MF * 2);
  // Reuse dead buffers:
  half_t* ctxh = xh;     // x (f16) dead after QKV projections
  half_t* hh   = Qh;     // Q dead after attention
  float*  tmpF = attnF;  // attn_out dead after LN1
  (void)ws_size; (void)n_in; (void)in_sizes; (void)out_size;

  // --- stage f16 operands ---
  cvt_f32_to_f16<<<dim3((MD + 255) / 256), dim3(256), 0, stream>>>(x, xh, (int)MD);
  transpose_cvt<<<dim3((DD * DD + 255) / 256), dim3(256), 0, stream>>>(Wq, Wqt, DD, DD);
  transpose_cvt<<<dim3((DD * DD + 255) / 256), dim3(256), 0, stream>>>(Wk, Wkt, DD, DD);
  transpose_cvt<<<dim3((DD * DD + 255) / 256), dim3(256), 0, stream>>>(Wv, Wvt, DD, DD);
  transpose_cvt<<<dim3((DD * DD + 255) / 256), dim3(256), 0, stream>>>(Wo, Wot, DD, DD);
  transpose_cvt<<<dim3((DD * DFF + 255) / 256), dim3(256), 0, stream>>>(W1, W1t, DD, DFF);
  transpose_cvt<<<dim3((DFF * DD + 255) / 256), dim3(256), 0, stream>>>(W2, W2t, DFF, DD);

  const dim3 gemmBlk(128);
  const dim3 gD(DD / 256, MTOT / 16);    // N=768  -> (3, 512)
  const dim3 gF(DFF / 256, MTOT / 16);   // N=1536 -> (6, 512)

  // --- QKV projections ---
  gemm_wmma<<<gD, gemmBlk, 0, stream>>>(xh, Wqt, bq, nullptr, nullptr, Qh,
                                        MTOT, DD, DD, 0);
  gemm_wmma<<<gD, gemmBlk, 0, stream>>>(xh, Wkt, bk, nullptr, nullptr, Kh,
                                        MTOT, DD, DD, 0);
  gemm_wmma<<<gD, gemmBlk, 0, stream>>>(xh, Wvt, bv, nullptr, nullptr, Vt,
                                        MTOT, DD, DD, 1);

  // --- attention ---
  const size_t attnLds = (size_t)16 * SS * 4 + (size_t)16 * SS * 2 + 64;
  attn_wmma<<<dim3(BB * HH * (SS / 16)), dim3(32), attnLds, stream>>>(
      Qh, Kh, Vt, msk, ctxh);

  // --- output projection + residual ---
  gemm_wmma<<<gD, gemmBlk, 0, stream>>>(ctxh, Wot, bo, x, attnF, nullptr,
                                        MTOT, DD, DD, 2);
  layernorm<<<dim3(MTOT), dim3(256), 0, stream>>>(attnF, g1, be1, hF, hh);

  // --- FFN ---
  gemm_wmma<<<gF, gemmBlk, 0, stream>>>(hh, W1t, b1, nullptr, nullptr, ff1h,
                                        MTOT, DFF, DD, 3);
  gemm_wmma<<<gD, gemmBlk, 0, stream>>>(ff1h, W2t, b2, hF, tmpF, nullptr,
                                        MTOT, DD, DFF, 2);
  layernorm<<<dim3(MTOT), dim3(256), 0, stream>>>(tmpF, g2, be2, out, nullptr);
}
